// GCN_18468359373370
// MI455X (gfx1250) — compile-verified, matrix operand order checked by reading, not run on hardware
//
#include <hip/hip_runtime.h>

typedef __bf16 bf16;
typedef __attribute__((ext_vector_type(16))) __bf16 v16bf;
typedef __attribute__((ext_vector_type(8)))  __bf16 v8bf;
typedef __attribute__((ext_vector_type(8)))  float  v8f;
typedef int v4i_vs __attribute__((vector_size(16)));  // matches builtin's pointee type

// ---- CDNA5 async global->LDS staging (ASYNCcnt path), with safe fallback ----
#if defined(__gfx1250__) && __has_builtin(__builtin_amdgcn_global_load_async_to_lds_b128)
#define HAVE_ASYNC_LDS 1
#endif

static __device__ __forceinline__ void stage16(const float* g, float* l) {
#if defined(HAVE_ASYNC_LDS)
  typedef __attribute__((address_space(1))) v4i_vs* g128p;  // global src
  typedef __attribute__((address_space(3))) v4i_vs* l128p;  // LDS dst
  __builtin_amdgcn_global_load_async_to_lds_b128((g128p)(g), (l128p)(l), 0, 0);
#else
  *(float4*)l = *(const float4*)g;  // sync fallback: load + ds_store
#endif
}

template <int PENDING>
static __device__ __forceinline__ void wait_async() {
#if defined(HAVE_ASYNC_LDS)
#if __has_builtin(__builtin_amdgcn_s_wait_asynccnt)
  __builtin_amdgcn_s_wait_asynccnt((unsigned short)PENDING);
#else
  asm volatile("s_wait_asynccnt %0" ::"i"(PENDING) : "memory");
#endif
#endif
}

// ---------------- prep kernels ----------------

__global__ void k_deg_init(float* deg, int n) {
  int i = blockIdx.x * blockDim.x + threadIdx.x;
  if (i < n) deg[i] = 1.0f;  // self-loop contribution
}

__global__ void k_deg_accum(const int* __restrict__ dst, int n_edges, float* deg) {
  int e = blockIdx.x * blockDim.x + threadIdx.x;
  if (e < n_edges) atomicAdd(&deg[dst[e]], 1.0f);
}

__global__ void k_dinv(const float* __restrict__ deg, float* __restrict__ dinv, int n) {
  int i = blockIdx.x * blockDim.x + threadIdx.x;
  if (i < n) {
    float d = deg[i];
    dinv[i] = d > 0.0f ? rsqrtf(d) : 0.0f;
  }
}

// W[K,N] row-major (f32) -> WT[N,K] row-major (bf16)
__global__ void k_transpose_cvt(const float* __restrict__ W, bf16* __restrict__ WT, int K, int N) {
  int idx = blockIdx.x * blockDim.x + threadIdx.x;
  if (idx < K * N) {
    int k = idx / N, n = idx % N;
    WT[(size_t)n * K + k] = (bf16)W[idx];
  }
}

// ---------------- WMMA GEMM: C[M,N] = (relu?)A_f32[M,K] @ BT_bf16[N,K]^T ----------------
// Block: N/16 waves, one 16-row M tile per block.
// A tile (16x32 f32) staged into LDS via double-buffered async DMA; relu +
// f32->bf16 conversion fused into the per-lane fragment load.
template <int K, int N, bool RELU_A>
__global__ void __launch_bounds__((N / 16) * 32) k_wmma_gemm(
    const float* __restrict__ A, const bf16* __restrict__ BT,
    float* __restrict__ C, int M) {
  static_assert(K % 32 == 0 && N % 16 == 0, "shape");
  __shared__ __align__(16) float Asf[2][16][32];  // double-buffered A tile (4 KB)

  const int lane  = threadIdx.x & 31;
  const int wave  = threadIdx.x >> 5;
  const int l15   = lane & 15;
  const int khalf = lane >> 4;
  const int m0    = blockIdx.x * 16;
  const int n0    = wave * 16;

  const bf16* bbase = BT + (size_t)(n0 + l15) * K + khalf * 8;

  // staging: 128 chunks of 16B cover the 16x32 f32 tile
  const int chunk = threadIdx.x;
  const int smi = chunk >> 3;         // tile row
  const int sco = (chunk & 7) * 4;    // col (floats)
  int srow = m0 + smi;
  if (srow > M - 1) srow = M - 1;
  const float* sgp = A + (size_t)srow * K + sco;

  auto issue = [&](int kk, int buf) {
    if (chunk < 128) stage16(sgp + kk, &Asf[buf][smi][sco]);
  };

  issue(0, 0);
  int cur = 0;
  v8f acc = {};
  for (int kk = 0; kk < K; kk += 32) {
    const bool last = (kk + 32 >= K);
    if (!last) {
      issue(kk + 32, cur ^ 1);  // prefetch next tile into other buffer
      wait_async<1>();          // current tile done (in-order), next may fly
    } else {
      wait_async<0>();
    }
    __syncthreads();  // publish DMA'd tile to all waves

    // A fragment per ISA layout: lane(l15)=M, khalf selects k-octet; two 32B runs
    const float* ap = &Asf[cur][l15][khalf * 8];
    float4 f0 = *(const float4*)(ap);
    float4 f1 = *(const float4*)(ap + 4);
    float4 f2 = *(const float4*)(ap + 16);
    float4 f3 = *(const float4*)(ap + 20);
    if (RELU_A) {
      f0.x = fmaxf(f0.x, 0.f); f0.y = fmaxf(f0.y, 0.f); f0.z = fmaxf(f0.z, 0.f); f0.w = fmaxf(f0.w, 0.f);
      f1.x = fmaxf(f1.x, 0.f); f1.y = fmaxf(f1.y, 0.f); f1.z = fmaxf(f1.z, 0.f); f1.w = fmaxf(f1.w, 0.f);
      f2.x = fmaxf(f2.x, 0.f); f2.y = fmaxf(f2.y, 0.f); f2.z = fmaxf(f2.z, 0.f); f2.w = fmaxf(f2.w, 0.f);
      f3.x = fmaxf(f3.x, 0.f); f3.y = fmaxf(f3.y, 0.f); f3.z = fmaxf(f3.z, 0.f); f3.w = fmaxf(f3.w, 0.f);
    }
    v16bf a;
    a[0]  = (bf16)f0.x; a[1]  = (bf16)f0.y; a[2]  = (bf16)f0.z; a[3]  = (bf16)f0.w;
    a[4]  = (bf16)f1.x; a[5]  = (bf16)f1.y; a[6]  = (bf16)f1.z; a[7]  = (bf16)f1.w;
    a[8]  = (bf16)f2.x; a[9]  = (bf16)f2.y; a[10] = (bf16)f2.z; a[11] = (bf16)f2.w;
    a[12] = (bf16)f3.x; a[13] = (bf16)f3.y; a[14] = (bf16)f3.z; a[15] = (bf16)f3.w;

    // B fragment from transposed weights: lane(l15)=N, same k pattern
    v8bf b0 = *(const v8bf*)(bbase + kk);
    v8bf b1 = *(const v8bf*)(bbase + kk + 16);
    v16bf b;
#pragma unroll
    for (int j = 0; j < 8; ++j) { b[j] = b0[j]; b[j + 8] = b1[j]; }

    acc = __builtin_amdgcn_wmma_f32_16x16x32_bf16(false, a, false, b, (short)0, acc,
                                                  false, false);
    __syncthreads();  // all reads of Asf[cur] done before it is re-filled
    cur ^= 1;
  }

  // C layout: VGPR i -> M = i + 8*khalf, N = l15
#pragma unroll
  for (int i = 0; i < 8; ++i) {
    int m = m0 + i + 8 * khalf;
    if (m < M) C[(size_t)m * N + n0 + l15] = acc[i];
  }
}

// ---------------- GCN aggregation ----------------

// H[i,c] = bias[c] + dinv[i]^2 * XW[i,c]   (self-loop term + bias)
template <int C>
__global__ void k_bias_selfloop(const float* __restrict__ XW, const float* __restrict__ dinv,
                                const float* __restrict__ bias, float* __restrict__ H,
                                int n_nodes) {
  long long idx = (long long)blockIdx.x * blockDim.x + threadIdx.x;
  if (idx >= (long long)n_nodes * C) return;
  int node = (int)(idx / C);
  int c = (int)(idx % C);
  float di = dinv[node];
  H[idx] = bias[c] + di * di * XW[idx];
}

// H[d] += dinv[s]*dinv[d] * XW[s]  (float4 per thread, float atomics)
template <int C>
__global__ void k_edge_scatter(const float* __restrict__ XW, const float* __restrict__ dinv,
                               const int* __restrict__ src, const int* __restrict__ dst,
                               int n_edges, float* __restrict__ H) {
  constexpr int G = C / 4;
  long long tid = (long long)blockIdx.x * blockDim.x + threadIdx.x;
  long long e = tid / G;
  if (e >= n_edges) return;
  int g = (int)(tid % G);
  int s = src[e], d = dst[e];
  float w = dinv[s] * dinv[d];
  float4 v = *(const float4*)(XW + (size_t)s * C + g * 4);
  float* o = H + (size_t)d * C + g * 4;
  atomicAdd(o + 0, w * v.x);
  atomicAdd(o + 1, w * v.y);
  atomicAdd(o + 2, w * v.z);
  atomicAdd(o + 3, w * v.w);
}

// ---------------- fused edge scorer ----------------
// out[l,:] = relu(relu(h2[s])*relu(h2[d]) @ fcW1 + fcb1) @ fcW2 + fcb2
// 4 waves/block, 16 label-edges per wave; WMMA on 16x64 e-tile vs 64x64 fcW1.
__global__ void __launch_bounds__(128) k_edge_score(
    const float* __restrict__ H2, const bf16* __restrict__ fcW1T,
    const float* __restrict__ fcb1, const float* __restrict__ fcW2,
    const float* __restrict__ fcb2, const int* __restrict__ lsrc,
    const int* __restrict__ ldst, float* __restrict__ out, int n_label) {
  __shared__ __align__(16) bf16 etile[4][16][64];   // 8 KB
  __shared__ __align__(16) float ytile[4][16][66];  // padded, ~16.5 KB

  const int lane  = threadIdx.x & 31;
  const int wave  = threadIdx.x >> 5;
  const int l15   = lane & 15;
  const int khalf = lane >> 4;
  const int e0    = (blockIdx.x * 4 + wave) * 16;

  // phase 1: gather pair embeddings (relu fused), elementwise product -> LDS bf16
  {
    int er = lane >> 1;
    int el = e0 + er;
    int c0 = (lane & 1) * 32;
    int eidx = el < n_label ? el : 0;
    int s = lsrc[eidx], d = ldst[eidx];
    const float4* ps = (const float4*)(H2 + (size_t)s * 64 + c0);
    const float4* pd = (const float4*)(H2 + (size_t)d * 64 + c0);
#pragma unroll
    for (int q = 0; q < 8; ++q) {
      float4 av = ps[q], bv = pd[q];
      bf16* t = &etile[wave][er][c0 + q * 4];
      t[0] = (bf16)(fmaxf(av.x, 0.0f) * fmaxf(bv.x, 0.0f));
      t[1] = (bf16)(fmaxf(av.y, 0.0f) * fmaxf(bv.y, 0.0f));
      t[2] = (bf16)(fmaxf(av.z, 0.0f) * fmaxf(bv.z, 0.0f));
      t[3] = (bf16)(fmaxf(av.w, 0.0f) * fmaxf(bv.w, 0.0f));
    }
  }
  __syncthreads();

  // phase 2: y[16,64] = e[16,64] @ fcW1[64,64]  (2 k-steps x 4 n-tiles of WMMA)
  v8f acc[4] = {};
#pragma unroll
  for (int ks = 0; ks < 64; ks += 32) {
    const bf16* ap = &etile[wave][l15][ks + khalf * 8];
    v8bf a0 = *(const v8bf*)ap;
    v8bf a1 = *(const v8bf*)(ap + 16);
    v16bf a;
#pragma unroll
    for (int j = 0; j < 8; ++j) { a[j] = a0[j]; a[j + 8] = a1[j]; }
#pragma unroll
    for (int t4 = 0; t4 < 4; ++t4) {
      const bf16* bp = fcW1T + (size_t)(16 * t4 + l15) * 64 + ks + khalf * 8;
      v8bf b0 = *(const v8bf*)bp;
      v8bf b1 = *(const v8bf*)(bp + 16);
      v16bf b;
#pragma unroll
      for (int j = 0; j < 8; ++j) { b[j] = b0[j]; b[j + 8] = b1[j]; }
      acc[t4] = __builtin_amdgcn_wmma_f32_16x16x32_bf16(false, a, false, b, (short)0,
                                                        acc[t4], false, false);
    }
  }

  // phase 3: bias + relu -> LDS f32
#pragma unroll
  for (int t4 = 0; t4 < 4; ++t4) {
    int nn = 16 * t4 + l15;
    float bias = fcb1[nn];
#pragma unroll
    for (int i = 0; i < 8; ++i) {
      int mm = i + 8 * khalf;
      ytile[wave][mm][nn] = fmaxf(acc[t4][i] + bias, 0.0f);
    }
  }
  __syncthreads();

  // phase 4: tiny 64->2 projection; lane = (edge in tile, output k)
  {
    int me = l15;
    int k = khalf;  // 0 or 1
    int el = e0 + me;
    float sum = fcb2[k];
#pragma unroll
    for (int n = 0; n < 64; ++n) sum += ytile[wave][me][n] * fcW2[n * 2 + k];
    if (el < n_label) out[(size_t)el * 2 + k] = sum;
  }
}

// ---------------- launcher ----------------

extern "C" void kernel_launch(void* const* d_in, const int* in_sizes, int n_in,
                              void* d_out, int out_size, void* d_ws, size_t ws_size,
                              hipStream_t stream) {
  const float* x    = (const float*)d_in[0];
  const float* W1   = (const float*)d_in[1];
  const float* b1   = (const float*)d_in[2];
  const float* W2   = (const float*)d_in[3];
  const float* b2   = (const float*)d_in[4];
  const float* fcW1 = (const float*)d_in[5];
  const float* fcb1 = (const float*)d_in[6];
  const float* fcW2 = (const float*)d_in[7];
  const float* fcb2 = (const float*)d_in[8];
  const int* eidx   = (const int*)d_in[9];
  const int* elidx  = (const int*)d_in[10];

  const int F_IN = 512, C1 = 128, C2 = 64;
  const int n_nodes = in_sizes[0] / F_IN;
  const int n_edges = in_sizes[9] / 2;
  const int n_label = in_sizes[10] / 2;
  const int* src  = eidx;
  const int* dst  = eidx + n_edges;
  const int* lsrc = elidx;
  const int* ldst = elidx + n_label;

  char* ws = (char*)d_ws;
  size_t off = 0;
  auto carve = [&](size_t bytes) -> char* {
    char* p = ws + off;
    off += (bytes + 255) & ~(size_t)255;
    return p;
  };
  bf16*  W1T   = (bf16*)carve((size_t)C1 * F_IN * sizeof(bf16));
  bf16*  W2T   = (bf16*)carve((size_t)C2 * C1 * sizeof(bf16));
  bf16*  fcW1T = (bf16*)carve((size_t)64 * 64 * sizeof(bf16));
  float* deg   = (float*)carve((size_t)n_nodes * sizeof(float));
  float* dinv  = (float*)carve((size_t)n_nodes * sizeof(float));
  float* bufA  = (float*)carve((size_t)n_nodes * C1 * sizeof(float));  // XW1, then XW2
  float* bufB  = (float*)carve((size_t)n_nodes * C1 * sizeof(float));  // H1, then H2
  (void)ws_size; (void)n_in; (void)out_size;

  const int T = 256;
  // degrees (shared by both layers) and D^-1/2
  k_deg_init<<<(n_nodes + T - 1) / T, T, 0, stream>>>(deg, n_nodes);
  k_deg_accum<<<(n_edges + T - 1) / T, T, 0, stream>>>(dst, n_edges, deg);
  k_dinv<<<(n_nodes + T - 1) / T, T, 0, stream>>>(deg, dinv, n_nodes);
  // weight prep: transpose + bf16 convert
  k_transpose_cvt<<<(F_IN * C1 + T - 1) / T, T, 0, stream>>>(W1, W1T, F_IN, C1);
  k_transpose_cvt<<<(C1 * C2 + T - 1) / T, T, 0, stream>>>(W2, W2T, C1, C2);
  k_transpose_cvt<<<(64 * 64 + T - 1) / T, T, 0, stream>>>(fcW1, fcW1T, 64, 64);

  const int mblocks = (n_nodes + 15) / 16;

  // layer 1: XW1 = x @ W1 ; H1 = b1 + dinv^2*XW1 + scatter(norm * XW1[src])
  k_wmma_gemm<512, 128, false><<<mblocks, 256, 0, stream>>>(x, W1T, bufA, n_nodes);
  long long tot1 = (long long)n_nodes * C1;
  k_bias_selfloop<128><<<(int)((tot1 + T - 1) / T), T, 0, stream>>>(bufA, dinv, b1, bufB, n_nodes);
  long long sc1 = (long long)n_edges * (C1 / 4);
  k_edge_scatter<128><<<(int)((sc1 + T - 1) / T), T, 0, stream>>>(bufA, dinv, src, dst, n_edges, bufB);

  // layer 2: XW2 = relu(H1) @ W2 (relu fused into A fragment load)
  k_wmma_gemm<128, 64, true><<<mblocks, 128, 0, stream>>>(bufB, W2T, bufA, n_nodes);
  long long tot2 = (long long)n_nodes * C2;
  k_bias_selfloop<64><<<(int)((tot2 + T - 1) / T), T, 0, stream>>>(bufA, dinv, b2, bufB, n_nodes);
  long long sc2 = (long long)n_edges * (C2 / 4);
  k_edge_scatter<64><<<(int)((sc2 + T - 1) / T), T, 0, stream>>>(bufA, dinv, src, dst, n_edges, bufB);

  // fused edge scorer (relu of H2 applied at gather time)
  const int sblocks = (n_label + 63) / 64;
  k_edge_score<<<sblocks, 128, 0, stream>>>(bufB, fcW1T, fcb1, fcW2, fcb2, lsrc, ldst,
                                            (float*)d_out, n_label);
}
